// MultiScaleAttention_6803228197120
// MI455X (gfx1250) — matrix operand order repaired and checked by reading, not compiled, not run on hardware
//
#include <hip/hip_runtime.h>
#include <hip/hip_bf16.h>
#include <math.h>

// ---------------------------------------------------------------------------
// MI455X (gfx1250) multi-scale attention block.
//
// Roofline: ~120 GFLOP of GEMM vs a few hundred MB of streamed activations
// (23.3 TB/s -> ~20us). bf16 WMMA (16x16x32, f32 accum) keeps the math time
// at the same order as the memory time; fp32 WMMA (K=4) would be ~8x slower.
// All fragment loads are vectorized to b128 via the ISA 7.12.2 layouts:
// A frags = two contiguous runs of 8 bf16 per lane; B frags = two contiguous
// runs of 8 bf16 per lane when K is the contiguous axis, so weights are
// stored transposed ([N][K]) and V is stored [head][d][key]. GEMM tile width
// is a template parameter so the inner loop is branch-free: one clause of
// b128 loads + 8 independent WMMA chains per K step.
// Attention keeps the 16xH score slab in LDS (wave32 shfl softmax): the
// O(H^2) matrix never reaches HBM.
// ---------------------------------------------------------------------------

typedef __bf16 bf16;
typedef __bf16 v16bf __attribute__((ext_vector_type(16)));
typedef __bf16 v8bf  __attribute__((ext_vector_type(8)));
typedef float  v8f   __attribute__((ext_vector_type(8)));

#define DIM   768
#define D4    192
#define HEADS 12
#define HD    64
#define BB    4
#define NN    1024

// ---- vectorized WMMA fragment loads (CDNA5 ISA 7.12.2, wave32) -------------
__device__ __forceinline__ v16bf cat8(v8bf lo, v8bf hi) {
    return __builtin_shufflevector(lo, hi, 0, 1, 2, 3, 4, 5, 6, 7,
                                           8, 9, 10, 11, 12, 13, 14, 15);
}
// A (16-bit 16x32): lane holds row M=lane%16; slots 0-7 = K base..base+7,
// slots 8-15 = K base+16..base+23, base = (lane>=16)*8.  rowp -> &X[m*K + k0]
__device__ __forceinline__ v16bf load_a_frag(const bf16* rowp, int lane) {
    int base = (lane >> 4) << 3;
    v8bf lo = *(const v8bf*)(rowp + base);
    v8bf hi = *(const v8bf*)(rowp + base + 16);
    return cat8(lo, hi);
}
// B (16-bit 32x16): lane holds col N=lane%16; slots 0-7 = K base..base+7,
// slots 8-15 = K base+16..base+23, base = (lane>=16)*8. p -> &Wt[n*K + k0] (K contiguous in memory).
__device__ __forceinline__ v16bf load_b_frag(const bf16* p, int lane) {
    const bf16* q = p + ((lane >> 4) << 3);
    v8bf lo = *(const v8bf*)(q);
    v8bf hi = *(const v8bf*)(q + 16);
    return cat8(lo, hi);
}
// C/D (f32 16x16): VGPR v, lane L -> row = v + (L/16)*8, col = L%16.

__device__ __forceinline__ float wave_sum(float v) {
    #pragma unroll
    for (int o = 16; o > 0; o >>= 1) v += __shfl_xor(v, o);
    return v;
}
__device__ __forceinline__ float wave_max(float v) {
    #pragma unroll
    for (int o = 16; o > 0; o >>= 1) v = fmaxf(v, __shfl_xor(v, o));
    return v;
}
__device__ __forceinline__ float gelu_f(float x) {
    return 0.5f * x * (1.0f + erff(x * 0.70710678118654752f));
}

// ---- conversions -----------------------------------------------------------
__global__ void cvt_f32_bf16_kernel(const float* __restrict__ s, bf16* __restrict__ d, int n) {
    int i = blockIdx.x * 256 + threadIdx.x;
    if (i < n) d[i] = (bf16)s[i];
}

// S slabs of [K x N] f32 row-major -> bf16 transposed [N x K] per slab.
__global__ void cvt_transpose_kernel(const float* __restrict__ W, bf16* __restrict__ Wt,
                                     int K, int N, int S) {
    int idx = blockIdx.x * 256 + threadIdx.x;
    if (idx >= S * K * N) return;
    int s = idx / (K * N);
    int r = idx % (K * N);
    int k = r / N, n = r % N;
    Wt[((size_t)s * N + n) * K + k] = (bf16)W[idx];
}

// linear interp along seq dim of [B, Hin, C] -> [B, Hout, C] (align_corners=False)
__global__ void interp_kernel(const float* __restrict__ src, float* __restrict__ dst,
                              int Hin, int Hout) {
    int idx = blockIdx.x * 256 + threadIdx.x;
    if (idx >= BB * Hout * DIM) return;
    int c = idx % DIM;
    int j = (idx / DIM) % Hout;
    int b = idx / (DIM * Hout);
    float s  = fmaxf((j + 0.5f) * ((float)Hin / (float)Hout) - 0.5f, 0.0f);
    int   i0 = (int)floorf(s);
    int   i1 = min(i0 + 1, Hin - 1);
    float t  = s - (float)i0;
    float v0 = src[((size_t)b * Hin + i0) * DIM + c];
    float v1 = src[((size_t)b * Hin + i1) * DIM + c];
    dst[idx] = v0 * (1.0f - t) + v1 * t;
}

// ---- bf16 WMMA GEMM: Y[MxN] = X[MxK] @ W[KxN] + bias -----------------------
// Wt is W transposed ([N][K], K contiguous). 128 threads = 4 waves; each wave
// owns a 16 x (NT*16) tile (A frag reused NT times). Branch-free inner loop;
// all frag loads are b128. Requires M%64==0, N%(NT*16)==0, K%32==0.
template <int NT>
__global__ __launch_bounds__(128)
void gemm_bf16_kernel(const bf16* __restrict__ X, const bf16* __restrict__ Wt,
                      const float* __restrict__ bias, float* __restrict__ Y,
                      int M, int N, int K) {
    int lane = threadIdx.x & 31;
    int wave = threadIdx.x >> 5;
    int m0 = blockIdx.x * 64 + wave * 16;
    int n0 = blockIdx.y * (NT * 16);

    v8f acc[NT] = {};
    const bf16* xrow = X + (size_t)(m0 + (lane & 15)) * K;
    const bf16* wrow = Wt + (size_t)(n0 + (lane & 15)) * K;

    for (int k0 = 0; k0 < K; k0 += 32) {
        v16bf a = load_a_frag(xrow + k0, lane);
        __builtin_prefetch(xrow + k0 + 32, 0, 1);   // global_prefetch_b8 (branch-free)
        #pragma unroll
        for (int t = 0; t < NT; t++) {
            v16bf b = load_b_frag(wrow + (size_t)t * 16 * K + k0, lane);
            acc[t] = __builtin_amdgcn_wmma_f32_16x16x32_bf16(
                false, a, false, b, (short)0, acc[t], false, false);
        }
    }
    #pragma unroll
    for (int t = 0; t < NT; t++) {
        int n = n0 + t * 16 + (lane & 15);
        float bv = bias ? bias[n] : 0.0f;
        #pragma unroll
        for (int vv = 0; vv < 8; vv++) {
            int r = m0 + vv + ((lane >> 4) << 3);
            Y[(size_t)r * N + n] = acc[t][vv] + bv;
        }
    }
}

// ---- LayerNorm + exact GELU over rows; optional bf16 mirror ----------------
__global__ __launch_bounds__(256)
void ln_gelu_kernel(float* __restrict__ Y, const float* __restrict__ g,
                    const float* __restrict__ be, bf16* __restrict__ Ybf, int C) {
    int row = blockIdx.x;
    float* yr = Y + (size_t)row * C;
    int tid = threadIdx.x, lane = tid & 31, wv = tid >> 5;
    __shared__ float sa[8], sb[8], st[2];
    float s = 0.f, s2 = 0.f;
    for (int c = tid; c < C; c += 256) { float x = yr[c]; s += x; s2 += x * x; }
    s = wave_sum(s); s2 = wave_sum(s2);
    if (lane == 0) { sa[wv] = s; sb[wv] = s2; }
    __syncthreads();
    if (tid == 0) {
        float ts = 0.f, t2 = 0.f;
        for (int i = 0; i < 8; i++) { ts += sa[i]; t2 += sb[i]; }
        float mu = ts / C;
        st[0] = mu; st[1] = t2 / C - mu * mu;
    }
    __syncthreads();
    float mu = st[0], inv = rsqrtf(st[1] + 1e-5f);
    for (int c = tid; c < C; c += 256) {
        float x  = (yr[c] - mu) * inv * g[c] + be[c];
        float ge = gelu_f(x);
        yr[c] = ge;
        if (Ybf) Ybf[(size_t)row * C + c] = (bf16)ge;
    }
}

// mean over seq dim of [B, H, C]; out[b*ostride + ooff + c]
__global__ void mean_seq_kernel(const float* __restrict__ X, float* __restrict__ out,
                                int H, int C, int ostride, int ooff) {
    int idx = blockIdx.x * 256 + threadIdx.x;
    if (idx >= BB * C) return;
    int b = idx / C, c = idx % C;
    float s = 0.f;
    for (int p = 0; p < H; p++) s += X[((size_t)b * H + p) * C + c];
    out[b * ostride + ooff + c] = s / (float)H;
}

// channel-attention gate: cw = sigmoid(gelu(ln(m@W1+b1)) @ W2 + b2); fp32 (tiny)
__global__ __launch_bounds__(256)
void ca_kernel(const float* __restrict__ m, const float* __restrict__ W1,
               const float* __restrict__ b1, const float* __restrict__ gg,
               const float* __restrict__ be, const float* __restrict__ W2,
               const float* __restrict__ b2, float* __restrict__ cw) {
    int b = blockIdx.x, tid = threadIdx.x, lane = tid & 31, wv = tid >> 5;
    __shared__ float hid[D4];
    __shared__ float sa[8], sb[8], st[2];
    const float* mb = m + b * DIM;
    if (tid < D4) {
        float s = b1[tid];
        for (int k = 0; k < DIM; k++) s += mb[k] * W1[k * D4 + tid];
        hid[tid] = s;
    }
    __syncthreads();
    float sv = (tid < D4) ? hid[tid] : 0.f;
    float sv2 = sv * sv;
    sv = wave_sum(sv); sv2 = wave_sum(sv2);
    if (lane == 0) { sa[wv] = sv; sb[wv] = sv2; }
    __syncthreads();
    if (tid == 0) {
        float ts = 0.f, t2 = 0.f;
        for (int i = 0; i < 8; i++) { ts += sa[i]; t2 += sb[i]; }
        float mu = ts / D4;
        st[0] = mu; st[1] = t2 / D4 - mu * mu;
    }
    __syncthreads();
    if (tid < D4) {
        float x = (hid[tid] - st[0]) * rsqrtf(st[1] + 1e-5f) * gg[tid] + be[tid];
        hid[tid] = gelu_f(x);
    }
    __syncthreads();
    for (int o = tid; o < DIM; o += 256) {
        float s = b2[o];
        for (int k = 0; k < D4; k++) s += hid[k] * W2[k * DIM + o];
        cw[b * DIM + o] = 1.0f / (1.0f + expf(-s));
    }
}

__global__ void scale_cw_kernel(float* __restrict__ xs, bf16* __restrict__ xsbf,
                                const float* __restrict__ cw, int H) {
    int idx = blockIdx.x * 256 + threadIdx.x;
    if (idx >= BB * H * DIM) return;
    int c = idx % DIM;
    int b = idx / (H * DIM);
    float v = xs[idx] * cw[b * DIM + c];
    xs[idx] = v;
    xsbf[idx] = (bf16)v;
}

// spatial gate tail: sw=sigmoid(t192 . W2 + b2) per row; xs *= sw (f32 + bf16)
__global__ __launch_bounds__(32)
void sa_tail_kernel(const float* __restrict__ t192, const float* __restrict__ W2,
                    const float* __restrict__ b2, float* __restrict__ xs,
                    bf16* __restrict__ xsbf) {
    int row = blockIdx.x, lane = threadIdx.x;
    const float* tr = t192 + (size_t)row * D4;
    float s = 0.f;
    for (int k = lane; k < D4; k += 32) s += tr[k] * W2[k];
    s = wave_sum(s);
    float sw = 1.0f / (1.0f + expf(-(s + b2[0])));
    float* xr = xs + (size_t)row * DIM;
    bf16*  xb = xsbf + (size_t)row * DIM;
    for (int c = lane; c < DIM; c += 32) {
        float v = xr[c] * sw;
        xr[c] = v;
        xb[c] = (bf16)v;
    }
}

// split fused qkv [B,H,2304] f32 -> bf16: q,k as [b,head][key][d]; v TRANSPOSED
// as [b,head][d][key] so PV B-fragments are contiguous b128 loads.
__global__ void split_qkv_kernel(const float* __restrict__ qkv, bf16* __restrict__ q,
                                 bf16* __restrict__ k, bf16* __restrict__ v, int H) {
    int idx = blockIdx.x * 256 + threadIdx.x;
    if (idx >= BB * H * 3 * DIM) return;
    int c = idx % (3 * DIM);
    int p = (idx / (3 * DIM)) % H;
    int b = idx / (3 * DIM * H);
    int which = c / DIM;
    int hc    = c % DIM;
    int head  = hc / HD, d = hc % HD;
    size_t bh = (size_t)b * HEADS + head;
    bf16 val = (bf16)qkv[idx];
    if (which == 0)      q[(bh * H + p) * HD + d] = val;
    else if (which == 1) k[(bh * H + p) * HD + d] = val;
    else                 v[(bh * HD + d) * H + p] = val;   // transposed
}

// ---- fused attention: one wave per (b, head, 16 q rows) --------------------
// scores (16xH f32) + probs (16xH bf16) live entirely in LDS (96*H bytes).
__global__ __launch_bounds__(32)
void attn_kernel(const bf16* __restrict__ q, const bf16* __restrict__ k,
                 const bf16* __restrict__ v, float* __restrict__ o, int H) {
    extern __shared__ char smem[];
    float* sc = (float*)smem;
    bf16*  pb = (bf16*)(smem + (size_t)16 * H * sizeof(float));
    int lane = threadIdx.x;
    int bh = blockIdx.x;                    // b*HEADS + h
    int b = bh / HEADS, h = bh % HEADS;
    int q0 = blockIdx.y * 16;
    const bf16* qb = q + (size_t)bh * H * HD;
    const bf16* kb = k + (size_t)bh * H * HD;
    const bf16* vt = v + (size_t)bh * HD * H;   // [d][key]

    // Q fragments (16x64 -> two 16x32 A frags), b128 loads
    const bf16* qrow = qb + (size_t)(q0 + (lane & 15)) * HD;
    v16bf aq0 = load_a_frag(qrow, lane);
    v16bf aq1 = load_a_frag(qrow + 32, lane);
    const float scale = 0.125f;  // HD^-0.5

    // S = Q K^T * scale + relative bias  -> LDS
    for (int kt = 0; kt < H / 16; kt++) {
        const bf16* krow = kb + (size_t)(kt * 16 + (lane & 15)) * HD;
        v16bf bk0 = load_b_frag(krow, lane);
        v16bf bk1 = load_b_frag(krow + 32, lane);
        v8f acc = {};
        acc = __builtin_amdgcn_wmma_f32_16x16x32_bf16(false, aq0, false, bk0, (short)0, acc, false, false);
        acc = __builtin_amdgcn_wmma_f32_16x16x32_bf16(false, aq1, false, bk1, (short)0, acc, false, false);
        #pragma unroll
        for (int vv = 0; vv < 8; vv++) {
            int r = vv + ((lane >> 4) << 3);
            int c = kt * 16 + (lane & 15);
            sc[r * H + c] = acc[vv] * scale + (float)((q0 + r) - c) * 0.02f;
        }
    }

    // row-wise softmax in LDS (wave32 reductions)
    for (int r = 0; r < 16; r++) {
        float mx = -1e30f;
        for (int c = lane; c < H; c += 32) mx = fmaxf(mx, sc[r * H + c]);
        mx = wave_max(mx);
        float s = 0.f;
        for (int c = lane; c < H; c += 32) {
            float e = expf(sc[r * H + c] - mx);
            sc[r * H + c] = e;
            s += e;
        }
        s = wave_sum(s);
        float inv = 1.0f / s;
        for (int c = lane; c < H; c += 32) pb[r * H + c] = (bf16)(sc[r * H + c] * inv);
    }

    // O = P V  (16xH @ Hx64); P A-frags via ds_load_b128, V B-frags contiguous
    v8f oacc[4] = {};
    for (int kt = 0; kt < H / 32; kt++) {
        v16bf ap = load_a_frag(pb + (size_t)(lane & 15) * H + kt * 32, lane);
        #pragma unroll
        for (int t = 0; t < 4; t++) {
            const bf16* vrow = vt + (size_t)(t * 16 + (lane & 15)) * H + kt * 32;
            v16bf bv = load_b_frag(vrow, lane);
            oacc[t] = __builtin_amdgcn_wmma_f32_16x16x32_bf16(false, ap, false, bv, (short)0, oacc[t], false, false);
        }
    }
    #pragma unroll
    for (int t = 0; t < 4; t++) {
        #pragma unroll
        for (int vv = 0; vv < 8; vv++) {
            int r = q0 + vv + ((lane >> 4) << 3);
            int c = h * HD + t * 16 + (lane & 15);
            o[((size_t)b * H + r) * DIM + c] = oacc[t][vv];
        }
    }
}

// fusion-weight softmax over 3 scales (tiny)
__global__ void fw_kernel(const float* __restrict__ cat, const float* __restrict__ W,
                          const float* __restrict__ bb, float* __restrict__ wsm) {
    __shared__ float lg[BB][3];
    int t = threadIdx.x;
    if (t < BB * 3) {
        int b = t / 3, j = t % 3;
        float s = bb[j];
        for (int c = 0; c < 3 * DIM; c++) s += cat[b * 3 * DIM + c] * W[c * 3 + j];
        lg[b][j] = s;
    }
    __syncthreads();
    if (t < BB) {
        float m = fmaxf(lg[t][0], fmaxf(lg[t][1], lg[t][2]));
        float e0 = expf(lg[t][0] - m), e1 = expf(lg[t][1] - m), e2 = expf(lg[t][2] - m);
        float s = e0 + e1 + e2;
        wsm[t * 3 + 0] = e0 / s; wsm[t * 3 + 1] = e1 / s; wsm[t * 3 + 2] = e2 / s;
    }
}

__global__ void combine_kernel(const float* __restrict__ f0, const float* __restrict__ f1,
                               const float* __restrict__ f2, const float* __restrict__ wsm,
                               bf16* __restrict__ xobf) {
    int idx = blockIdx.x * 256 + threadIdx.x;
    if (idx >= BB * NN * DIM) return;
    int b = idx / (NN * DIM);
    float v = f0[idx] * wsm[b * 3 + 0] + f1[idx] * wsm[b * 3 + 1] + f2[idx] * wsm[b * 3 + 2];
    xobf[idx] = (bf16)v;
}

__global__ void residual_kernel(float* __restrict__ out, const float* __restrict__ x) {
    int idx = blockIdx.x * 256 + threadIdx.x;
    if (idx < BB * NN * DIM) out[idx] += x[idx];
}

// ---------------------------------------------------------------------------
static inline int nb256(int n) { return (n + 255) / 256; }

extern "C" void kernel_launch(void* const* d_in, const int* in_sizes, int n_in,
                              void* d_out, int out_size, void* d_ws, size_t ws_size,
                              hipStream_t stream) {
    (void)in_sizes; (void)n_in; (void)out_size; (void)ws_size;
    const float* x      = (const float*)d_in[0];
    const float* fp_W   = (const float*)d_in[1];
    const float* fp_b   = (const float*)d_in[2];
    const float* fp_g   = (const float*)d_in[3];
    const float* fp_be  = (const float*)d_in[4];
    const float* ca_W1  = (const float*)d_in[5];
    const float* ca_b1  = (const float*)d_in[6];
    const float* ca_g   = (const float*)d_in[7];
    const float* ca_be  = (const float*)d_in[8];
    const float* ca_W2  = (const float*)d_in[9];
    const float* ca_b2  = (const float*)d_in[10];
    const float* sa_W1  = (const float*)d_in[11];
    const float* sa_b1  = (const float*)d_in[12];
    const float* sa_g   = (const float*)d_in[13];
    const float* sa_be  = (const float*)d_in[14];
    const float* sa_W2  = (const float*)d_in[15];
    const float* sa_b2  = (const float*)d_in[16];
    const float* qkv_W  = (const float*)d_in[17];
    const float* qkv_b  = (const float*)d_in[18];
    const float* fe_W1  = (const float*)d_in[19];
    const float* fe_b1  = (const float*)d_in[20];
    const float* fe_g   = (const float*)d_in[21];
    const float* fe_be  = (const float*)d_in[22];
    const float* fe_W2  = (const float*)d_in[23];
    const float* fe_b2  = (const float*)d_in[24];
    const float* fw_W   = (const float*)d_in[25];
    const float* fw_b   = (const float*)d_in[26];
    const float* proj_W = (const float*)d_in[27];
    const float* proj_b = (const float*)d_in[28];
    float* out = (float*)d_out;

    char* base = (char*)d_ws;
    size_t off = 0;
    auto take = [&](size_t bytes) -> char* {
        char* p = base + off;
        off += (bytes + 255) & ~(size_t)255;
        return p;
    };
    // bf16 weight pool (all stored TRANSPOSED: [N][K])
    bf16* wfp   = (bf16*)take((size_t)3 * DIM * DIM * 2);
    bf16* wsa1  = (bf16*)take((size_t)3 * DIM * D4 * 2);
    bf16* wqkv  = (bf16*)take((size_t)3 * DIM * 3 * DIM * 2);
    bf16* wfe1  = (bf16*)take((size_t)3 * DIM * 2 * DIM * 2);
    bf16* wfe2  = (bf16*)take((size_t)3 * 2 * DIM * DIM * 2);
    bf16* wproj = (bf16*)take((size_t)DIM * DIM * 2);
    // activations (with lifetime-based reuse)
    float* xs   = (float*)take((size_t)BB * NN * DIM * 4);        // xs / attn-out / fe2-out
    bf16*  xsbf = (bf16*) take((size_t)BB * NN * DIM * 2);        // xs / o / xo in bf16
    float* big  = (float*)take((size_t)BB * NN * 3 * DIM * 4);    // qkv f32, then fe hidden
    bf16*  bigbf= (bf16*) take((size_t)BB * NN * 3 * DIM * 2);    // q/k/v bf16, then fe hidden bf16
    float* t192 = (float*)take((size_t)BB * NN * D4 * 4);
    float* feats= (float*)take((size_t)3 * BB * NN * DIM * 4);
    float* mbuf = (float*)take((size_t)BB * DIM * 4);
    float* cw   = (float*)take((size_t)BB * DIM * 4);
    float* catb = (float*)take((size_t)BB * 3 * DIM * 4);
    float* wsm  = (float*)take((size_t)BB * 3 * 4);

    // --- convert weights to transposed bf16 (each call; no caching allowed) ---
    cvt_transpose_kernel<<<nb256(3 * DIM * DIM), 256, 0, stream>>>(fp_W, wfp, DIM, DIM, 3);
    cvt_transpose_kernel<<<nb256(3 * DIM * D4), 256, 0, stream>>>(sa_W1, wsa1, DIM, D4, 3);
    cvt_transpose_kernel<<<nb256(3 * DIM * 3 * DIM), 256, 0, stream>>>(qkv_W, wqkv, DIM, 3 * DIM, 3);
    cvt_transpose_kernel<<<nb256(3 * DIM * 2 * DIM), 256, 0, stream>>>(fe_W1, wfe1, DIM, 2 * DIM, 3);
    cvt_transpose_kernel<<<nb256(3 * 2 * DIM * DIM), 256, 0, stream>>>(fe_W2, wfe2, 2 * DIM, DIM, 3);
    cvt_transpose_kernel<<<nb256(DIM * DIM), 256, 0, stream>>>(proj_W, wproj, DIM, DIM, 1);

    const int Hs[3] = {1024, 768, 512};
    for (int i = 0; i < 3; i++) {
        int H = Hs[i], M = BB * H, E = M * DIM;
        bf16* qb = bigbf;
        bf16* kb = bigbf + (size_t)BB * HEADS * H * HD;
        bf16* vb = kb    + (size_t)BB * HEADS * H * HD;

        interp_kernel<<<nb256(E), 256, 0, stream>>>(x, xs, NN, H);
        cvt_f32_bf16_kernel<<<nb256(E), 256, 0, stream>>>(xs, xsbf, E);
        // fp projection + LN + GELU
        gemm_bf16_kernel<8><<<dim3(M / 64, DIM / 128), 128, 0, stream>>>(
            xsbf, wfp + (size_t)i * DIM * DIM, fp_b + i * DIM, xs, M, DIM, DIM);
        ln_gelu_kernel<<<M, 256, 0, stream>>>(xs, fp_g + i * DIM, fp_be + i * DIM, xsbf, DIM);
        // channel attention gate
        mean_seq_kernel<<<nb256(BB * DIM), 256, 0, stream>>>(xs, mbuf, H, DIM, DIM, 0);
        ca_kernel<<<BB, 256, 0, stream>>>(mbuf, ca_W1 + (size_t)i * DIM * D4, ca_b1 + i * D4,
                                          ca_g + i * D4, ca_be + i * D4,
                                          ca_W2 + (size_t)i * D4 * DIM, ca_b2 + i * DIM, cw);
        scale_cw_kernel<<<nb256(E), 256, 0, stream>>>(xs, xsbf, cw, H);
        // spatial attention gate (N=192 -> NT=4 tile, grid.y = 192/64 = 3)
        gemm_bf16_kernel<4><<<dim3(M / 64, D4 / 64), 128, 0, stream>>>(
            xsbf, wsa1 + (size_t)i * DIM * D4, sa_b1 + i * D4, t192, M, D4, DIM);
        ln_gelu_kernel<<<M, 256, 0, stream>>>(t192, sa_g + i * D4, sa_be + i * D4, nullptr, D4);
        sa_tail_kernel<<<M, 32, 0, stream>>>(t192, sa_W2 + (size_t)i * D4, sa_b2 + i, xs, xsbf);
        // qkv projection + attention
        gemm_bf16_kernel<8><<<dim3(M / 64, (3 * DIM) / 128), 128, 0, stream>>>(
            xsbf, wqkv + (size_t)i * DIM * 3 * DIM, qkv_b + i * 3 * DIM, big, M, 3 * DIM, DIM);
        split_qkv_kernel<<<nb256(M * 3 * DIM), 256, 0, stream>>>(big, qb, kb, vb, H);
        attn_kernel<<<dim3(BB * HEADS, H / 16), 32, (size_t)96 * H, stream>>>(qb, kb, vb, xs, H);
        // feature expansion (fe1 -> LN+GELU -> fe2)
        cvt_f32_bf16_kernel<<<nb256(E), 256, 0, stream>>>(xs, xsbf, E);
        gemm_bf16_kernel<8><<<dim3(M / 64, (2 * DIM) / 128), 128, 0, stream>>>(
            xsbf, wfe1 + (size_t)i * DIM * 2 * DIM, fe_b1 + i * 2 * DIM, big, M, 2 * DIM, DIM);
        ln_gelu_kernel<<<M, 256, 0, stream>>>(big, fe_g + i * 2 * DIM, fe_be + i * 2 * DIM, bigbf, 2 * DIM);
        gemm_bf16_kernel<8><<<dim3(M / 64, DIM / 128), 128, 0, stream>>>(
            bigbf, wfe2 + (size_t)i * 2 * DIM * DIM, fe_b2 + i * DIM, xs, M, DIM, 2 * DIM);
        // back to full resolution + per-scale mean for fusion weights
        float* fi = feats + (size_t)i * BB * NN * DIM;
        interp_kernel<<<nb256(BB * NN * DIM), 256, 0, stream>>>(xs, fi, H, NN);
        mean_seq_kernel<<<nb256(BB * DIM), 256, 0, stream>>>(fi, catb, NN, DIM, 3 * DIM, i * DIM);
    }

    // fusion weights, combine, output projection, residual
    fw_kernel<<<1, 64, 0, stream>>>(catb, fw_W, fw_b, wsm);
    combine_kernel<<<nb256(BB * NN * DIM), 256, 0, stream>>>(
        feats, feats + (size_t)BB * NN * DIM, feats + (size_t)2 * BB * NN * DIM, wsm, xsbf);
    gemm_bf16_kernel<8><<<dim3((BB * NN) / 64, DIM / 128), 128, 0, stream>>>(
        xsbf, wproj, proj_b, out, BB * NN, DIM, DIM);
    residual_kernel<<<nb256(BB * NN * DIM), 256, 0, stream>>>(out, x);
}
